// SelfAttention_26869315404643
// MI455X (gfx1250) — matrix operand order repaired and checked by reading, not compiled, not run on hardware
//
#include <hip/hip_runtime.h>
#include <hip/hip_bf16.h>

// ---------------- problem constants ----------------
#define BB   46
#define SS   50
#define DD   768
#define BS   (BB*SS)    // 2300 rows
#define BSP  2304       // padded to 16
#define MT   (BSP/16)   // 144 M tiles
#define NT   (DD/16)    // 48 N tiles
#define GAMA 0.96875f

// recurrence column split: 4 blocks per batch, 192 cols each
#define CSPLIT 4
#define CPB    (DD/CSPLIT)   // 192 columns per block
#define SROW   (CPB + 8)     // 200-float padded LDS row stride (8 mod 64 -> conflict-free)

typedef __attribute__((ext_vector_type(8)))  float        v8f;
typedef __attribute__((ext_vector_type(2)))  float        v2f;
typedef __attribute__((ext_vector_type(16))) __bf16       v16bf;
typedef __attribute__((ext_vector_type(4)))  unsigned int u32x4;

union FragBF { u32x4 q[2]; v16bf v; };

// ---------------- prep kernels ----------------
__global__ void k_pe(float* __restrict__ pe) {
    int i = blockIdx.x * 256 + threadIdx.x;
    if (i >= BB * DD) return;
    int b = i / DD, d = i % DD;
    // div_term for dim d uses value (d & ~1) == 2*j
    float div = __expf((float)(d & ~1) * (-9.210340371976184f / (float)DD));
    float ang = (float)b * div;
    pe[i] = (d & 1) ? __cosf(ang) : __sinf(ang);
}

__global__ void k_cvtA(const float* __restrict__ text, const float* __restrict__ emb,
                       __bf16* __restrict__ At, __bf16* __restrict__ Ae) {
    int i = blockIdx.x * 256 + threadIdx.x;
    if (i >= BSP * DD) return;
    bool in = (i < BS * DD);
    At[i] = in ? (__bf16)text[i] : (__bf16)0.0f;
    Ae[i] = in ? (__bf16)emb[i]  : (__bf16)0.0f;
}

__global__ void k_cvtW(const float* __restrict__ Wq, const float* __restrict__ Wk,
                       const float* __restrict__ Wv, __bf16* __restrict__ Wb) {
    int i = blockIdx.x * 256 + threadIdx.x;
    if (i >= DD * DD) return;
    Wb[i]             = (__bf16)Wq[i];
    Wb[i +   DD*DD]   = (__bf16)Wk[i];
    Wb[i + 2*DD*DD]   = (__bf16)Wv[i];
}

// ---------------- QKV projection: bf16 WMMA, f32 accumulate ----------------
// Y = X @ W^T + bias (+ pe[batch] for Q,K).  One wave per 16x16 output tile.
__global__ void k_qkv(const __bf16* __restrict__ At, const __bf16* __restrict__ Ae,
                      const __bf16* __restrict__ Wb,
                      const float* __restrict__ bq, const float* __restrict__ bk,
                      const float* __restrict__ bv, const float* __restrict__ pe,
                      float* __restrict__ Q, float* __restrict__ K, float* __restrict__ V) {
    int tm = blockIdx.x, tn = blockIdx.y, z = blockIdx.z;
    int lane = threadIdx.x & 31;
    int M = lane & 15, h = lane >> 4;

    const __bf16* A = (z == 0) ? At : Ae;
    const __bf16* W = Wb + (size_t)z * DD * DD;
    const float* bias = (z == 0) ? bq : (z == 1) ? bk : bv;
    float* Out = (z == 0) ? Q : (z == 1) ? K : V;

    const __bf16* arow = A + (size_t)(tm * 16 + M) * DD;   // A row for this lane
    const __bf16* wrow = W + (size_t)(tn * 16 + M) * DD;   // B col N == W row N

    v8f acc = {};
    for (int kk = 0; kk < DD / 32; ++kk) {
        int kb = kk * 32;
        FragBF fa, fb;
        // A 16x32 bf16 frag: halves K = kb+8h..+7 and kb+16+8h..+7
        const __bf16* ap = arow + kb + h * 8;
        fa.q[0] = *(const u32x4*)(ap);
        fa.q[1] = *(const u32x4*)(ap + 16);
        // B 32x16 bf16 frag: lane N holds K = kb+8h..+7 and kb+16+8h..+7
        const __bf16* wp = wrow + kb + h * 8;
        fb.q[0] = *(const u32x4*)(wp);
        fb.q[1] = *(const u32x4*)(wp + 16);
        acc = __builtin_amdgcn_wmma_f32_16x16x32_bf16(
            false, fa.v, false, fb.v, (short)0, acc, false, false);
    }

    int col = tn * 16 + M;
    float bval = bias[col];
#pragma unroll
    for (int r = 0; r < 8; ++r) {
        int row = tm * 16 + r + 8 * h;
        if (row < BS) {
            float v = acc[r] + bval;
            if (z < 2) v += pe[(row / SS) * DD + col];   // pe indexed by batch
            Out[(size_t)row * DD + col] = v;
        }
    }
}

// ---------------- attention (scores*8, softmax, ctx) + Q_trans ----------------
__global__ void k_attn(const float* __restrict__ Q, const float* __restrict__ Kb,
                       const float* __restrict__ V, const float* __restrict__ lin_w,
                       const float* __restrict__ lin_b,
                       float* __restrict__ Qt, float* __restrict__ ctx) {
    int q = blockIdx.x, b = blockIdx.y;
    int tid = threadIdx.x;
    __shared__ float qrow[DD];
    __shared__ float sc[64];

    const float* Qr = Q + ((size_t)b * SS + q) * DD;
    for (int d = tid; d < DD; d += 256) qrow[d] = Qr[d];
    __syncthreads();

    if (tid < SS) {
        const float* Kr = Kb + ((size_t)b * SS + tid) * DD;
        const float* Lr = lin_w + (size_t)tid * DD;
        float dk = 0.f, dl = 0.f;
        for (int d = 0; d < DD; ++d) {
            float qv = qrow[d];
            dk = fmaf(qv, Kr[d], dk);
            dl = fmaf(qv, Lr[d], dl);
        }
        sc[tid] = dk * 8.0f;
        Qt[((size_t)b * SS + q) * SS + tid] = dl + lin_b[tid];
    }
    __syncthreads();

    if (tid == 0) {
        float m = sc[0];
        for (int k = 1; k < SS; ++k) m = fmaxf(m, sc[k]);
        float s = 0.f;
        for (int k = 0; k < SS; ++k) { float e = __expf(sc[k] - m); sc[k] = e; s += e; }
        float inv = 1.0f / s;
        for (int k = 0; k < SS; ++k) sc[k] *= inv;
    }
    __syncthreads();

    for (int d = tid; d < DD; d += 256) {
        float a = 0.f;
        for (int k = 0; k < SS; ++k)
            a = fmaf(sc[k], V[((size_t)b * SS + k) * DD + d], a);
        ctx[((size_t)b * SS + q) * DD + d] = a;
    }
}

// ---------------- recurrence: s_i = (gamma*A) s_{i-1} + ctx[i], f32 WMMA ----
// Grid (BB, CSPLIT): block (b,c) owns 192 state columns for batch b.
// State (64 x 192, rows padded) lives in LDS, column-partitioned across the
// 4 waves (3 N-tiles of 16 cols each) -> no block barriers inside the 46-step
// loop (per-wave DS ops are in-order, columns are wave-private).
// ~66 KB LDS/block -> 4 blocks co-resident per 320 KB WGP; 184 blocks spread
// the 325 MB rec store stream across enough WGPs to reach the 23.3 TB/s HBM
// roofline.
__global__ void __launch_bounds__(128)
k_rec(const float* __restrict__ Qt, const float* __restrict__ ctx,
      float* __restrict__ rec) {
    extern __shared__ float lds[];
    float* Sst = lds;               // 64 * SROW state (local columns)
    float* Al  = lds + 64 * SROW;   // 64 * 64 gamma*Q_trans[b], zero-padded

    int b = blockIdx.x;
    int c = blockIdx.y;             // column-split index
    int tid = threadIdx.x;

    for (int i = tid; i < 64 * 64; i += 128) {
        int r = i >> 6, cc = i & 63;
        float v = 0.f;
        if (r < SS && cc < SS) v = GAMA * Qt[((size_t)b * SS + r) * SS + cc];
        Al[i] = v;
    }
    for (int i = tid; i < 64 * SROW; i += 128) Sst[i] = 0.f;
    __syncthreads();

    int lane = tid & 31, wave = tid >> 5;
    int M = lane & 15, h = lane >> 4;
    int colbase = c * CPB;          // this block's first global column

    for (int i = 0; i < BB; ++i) {
        const float* C = ctx + (size_t)i * SS * DD;
        float* R = rec + ((size_t)i * BB + b) * SS * DD;

        if (i + 1 < BB) {           // warm next step's ctx slice into cache
            const float* Cn = ctx + (size_t)(i + 1) * SS * DD;
            __builtin_prefetch(&Cn[(size_t)lane        * DD + colbase + wave * 48], 0, 0);
            __builtin_prefetch(&Cn[(size_t)(lane + 18) * DD + colbase + wave * 48], 0, 0);
        }

        for (int nn = 0; nn < 3; ++nn) {
            int lc  = (wave * 3 + nn) * 16 + M;   // local column (LDS)
            int col = colbase + lc;               // global column
            v8f acc[4];
#pragma unroll
            for (int m = 0; m < 4; ++m) {
                v8f a;
#pragma unroll
                for (int r = 0; r < 8; ++r) {
                    int row = m * 16 + r + 8 * h;
                    a[r] = (row < SS) ? C[(size_t)row * DD + col] : 0.0f;
                }
                const float* arow = Al + (m * 16 + M) * 64;
#pragma unroll
                for (int kk = 0; kk < 16; ++kk) {
                    v2f af, bf2;
                    af[0]  = arow[kk * 4 + 2 * h];
                    af[1]  = arow[kk * 4 + 2 * h + 1];
                    bf2[0] = Sst[(size_t)(kk * 4 + 2 * h)     * SROW + lc];
                    bf2[1] = Sst[(size_t)(kk * 4 + 2 * h + 1) * SROW + lc];
                    a = __builtin_amdgcn_wmma_f32_16x16x4_f32(
                        false, af, false, bf2, (short)0, a, false, false);
                }
                acc[m] = a;
            }
            // all reads of this wave's columns are done -> write back + emit rec
#pragma unroll
            for (int m = 0; m < 4; ++m) {
#pragma unroll
                for (int r = 0; r < 8; ++r) {
                    int row = m * 16 + r + 8 * h;
                    if (row < SS) R[(size_t)row * DD + col] = acc[m][r];
                    Sst[(size_t)row * SROW + lc] = acc[m][r];
                }
            }
        }
    }
}

// ---------------- host ----------------
extern "C" void kernel_launch(void* const* d_in, const int* in_sizes, int n_in,
                              void* d_out, int out_size, void* d_ws, size_t ws_size,
                              hipStream_t stream) {
    const float* text = (const float*)d_in[0];
    const float* emb  = (const float*)d_in[1];
    const float* Wq_w = (const float*)d_in[2];
    const float* Wq_b = (const float*)d_in[3];
    const float* Wk_w = (const float*)d_in[4];
    const float* Wk_b = (const float*)d_in[5];
    const float* Wv_w = (const float*)d_in[6];
    const float* Wv_b = (const float*)d_in[7];
    const float* lin_w = (const float*)d_in[8];
    const float* lin_b = (const float*)d_in[9];

    char* w = (char*)d_ws;
    size_t off = 0;
    auto take = [&](size_t bytes) -> void* {
        void* p = w + off;
        off += (bytes + 255) & ~(size_t)255;
        return p;
    };
    __bf16* At = (__bf16*)take((size_t)BSP * DD * 2);
    __bf16* Ae = (__bf16*)take((size_t)BSP * DD * 2);
    __bf16* Wb = (__bf16*)take((size_t)3 * DD * DD * 2);
    float*  pe = (float*) take((size_t)BB * DD * 4);
    float*  Q  = (float*) take((size_t)BS * DD * 4);
    float*  K  = (float*) take((size_t)BS * DD * 4);
    float*  V  = (float*) take((size_t)BS * DD * 4);
    float*  Qt = (float*) take((size_t)BB * SS * SS * 4);

    float* ctx = (float*)d_out;                      // [B,S,D]
    float* rec = ctx + (size_t)BB * SS * DD;         // [B,B,S,D]

    k_pe  <<<(BB * DD + 255) / 256, 256, 0, stream>>>(pe);
    k_cvtA<<<(BSP * DD + 255) / 256, 256, 0, stream>>>(text, emb, At, Ae);
    k_cvtW<<<(DD * DD + 255) / 256, 256, 0, stream>>>(Wq_w, Wk_w, Wv_w, Wb);

    k_qkv<<<dim3(MT, NT, 3), 32, 0, stream>>>(At, Ae, Wb, Wq_b, Wk_b, Wv_b, pe, Q, K, V);
    k_attn<<<dim3(SS, BB), 256, 0, stream>>>(Q, K, V, lin_w, lin_b, Qt, ctx);

    size_t recLds = (size_t)(64 * SROW + 64 * 64) * sizeof(float);  // ~66 KB
    k_rec<<<dim3(BB, CSPLIT), 128, recLds, stream>>>(Qt, ctx, rec);
}
